// GRUStack_18391049961748
// MI455X (gfx1250) — compile-verified, matrix operand order checked by reading, not run on hardware
//
#include <hip/hip_runtime.h>
#include <hip/hip_bf16.h>
#include <stdint.h>

// ---------------------------------------------------------------------------
// Types for CDNA5 WMMA (wave32): v_wmma_f32_16x16x32_bf16
// ---------------------------------------------------------------------------
typedef __attribute__((ext_vector_type(16))) __bf16 v16bf;
typedef __attribute__((ext_vector_type(8)))  float  v8f;

union Frag16 { uint4 q[2]; v16bf v; };

__device__ __forceinline__ unsigned short f32_to_bf16_rne(float f) {
  unsigned u = __float_as_uint(f);
  unsigned r = u + 0x7FFFu + ((u >> 16) & 1u);   // round-to-nearest-even
  return (unsigned short)(r >> 16);
}

// ---------------------------------------------------------------------------
// CDNA5 async memory->LDS copy (tracked with ASYNCcnt, no VGPR round-trip).
// Inline asm keeps it portable across ROCm7.2 / amdgpu-toolchain builtins.
// LDS destination = byte offset (low 32 bits of the generic pointer hold the
// LDS offset per the flat-aperture rules); source = 64-bit global address.
// ---------------------------------------------------------------------------
__device__ __forceinline__ void async_copy_b128(unsigned lds_byte_off,
                                                const void* g) {
  asm volatile("global_load_async_to_lds_b128 %0, %1, off"
               :: "v"(lds_byte_off), "v"((unsigned long long)(uintptr_t)g)
               : "memory");
}
__device__ __forceinline__ void wait_async0() {
  asm volatile("s_wait_asynccnt 0" ::: "memory");
}

// ---------------------------------------------------------------------------
// Elementwise fp32 -> bf16 conversion (weights / layer inputs)
// ---------------------------------------------------------------------------
__global__ void cvt_f32_bf16(const float* __restrict__ s,
                             unsigned short* __restrict__ d, long long n) {
  long long i = (long long)blockIdx.x * blockDim.x + threadIdx.x;
  long long stride = (long long)gridDim.x * blockDim.x;
  for (; i < n; i += stride) d[i] = f32_to_bf16_rne(s[i]);
}

// ---------------------------------------------------------------------------
// Stage one 128x32 bf16 A tile (8 KB = 512 x 16B) into LDS asynchronously.
// ---------------------------------------------------------------------------
__device__ __forceinline__ void stage_A_tile(const unsigned short* __restrict__ Abf,
                                             size_t rowbase, int K, int k0,
                                             unsigned lds_base) {
  for (int v = threadIdx.x; v < 512; v += 256) {
    int row = v >> 2, cu = v & 3;
    async_copy_b128(lds_base + (unsigned)v * 16u,
                    Abf + rowbase + (size_t)row * K + k0 + cu * 8);
  }
}

// ---------------------------------------------------------------------------
// Input-projection GEMM:  Xg[M,N] = Abf[M,K] * Wbf[N,K]^T + bias[N]
// Block tile: 128(M) x 64(N), 8 waves, each wave = one 16-row strip x 4 C tiles.
// A staged via async-to-LDS with double buffering (copy overlaps WMMA);
// B streamed from global (row of W contiguous in K => coalesced 16B loads
// matching the WMMA K-split register layout).
// ---------------------------------------------------------------------------
__global__ __launch_bounds__(256) void gemm_xg(
    const unsigned short* __restrict__ Abf,
    const unsigned short* __restrict__ Wbf,
    const float* __restrict__ bias,
    float* __restrict__ Xg, int M, int N, int K)
{
  __shared__ unsigned short As[2 * 128 * 32];    // double-buffered, 16 KB

  const int nblock = blockIdx.x * 64;
  const int mblock = blockIdx.y * 128;
  const int wave   = threadIdx.x >> 5;
  const int lane   = threadIdx.x & 31;
  const int lhalf  = lane >> 4;                  // K-half select
  const int ml     = lane & 15;                  // M (A) / N (B,C) index
  const int kb     = lhalf ? 8 : 0;              // K sub-offset within tile

  const unsigned lds_base = (unsigned)(uintptr_t)(&As[0]);
  const size_t  rowbase   = (size_t)mblock * K;
  const int     KT        = K >> 5;              // # of 32-wide K tiles

  v8f acc[4] = {v8f{}, v8f{}, v8f{}, v8f{}};

  // prologue: stage tile 0 into buffer 0
  stage_A_tile(Abf, rowbase, K, 0, lds_base);
  wait_async0();
  __syncthreads();

  for (int kt = 0; kt < KT; ++kt) {
    const int buf = kt & 1;
    const int k0  = kt << 5;

    // overlap: stage next K tile into the other buffer while computing
    if (kt + 1 < KT)
      stage_A_tile(Abf, rowbase, K, k0 + 32, lds_base + (buf ^ 1) * 8192u);

    Frag16 af;
    const uint4* ap =
        (const uint4*)(As + buf * 4096 + (wave * 16 + ml) * 32 + kb);
    af.q[0] = ap[0]; af.q[1] = ap[2];            // K 0..7 / 16..23 (or +8)

    // load all four B fragments first, then issue the four WMMAs
    Frag16 bfr[4];
#pragma unroll
    for (int j = 0; j < 4; ++j) {
      const unsigned short* wrow =
          Wbf + (size_t)(nblock + j * 16 + ml) * K + k0 + kb;
      __builtin_prefetch(wrow + 256, 0, 0);
      const uint4* bp = (const uint4*)wrow;
      bfr[j].q[0] = bp[0]; bfr[j].q[1] = bp[2];
    }
#pragma unroll
    for (int j = 0; j < 4; ++j)
      acc[j] = __builtin_amdgcn_wmma_f32_16x16x32_bf16(
          false, af.v, false, bfr[j].v, (short)0, acc[j], false, false);

    wait_async0();                               // next tile resident in LDS
    __syncthreads();                             // and everyone done reading
  }

#pragma unroll
  for (int j = 0; j < 4; ++j) {
    int n = nblock + j * 16 + ml;
    float bv = bias[n];
#pragma unroll
    for (int r = 0; r < 8; ++r) {
      int m = mblock + wave * 16 + r + lhalf * 8;
      Xg[(size_t)m * N + n] = acc[j][r] + bv;
    }
  }
}

// ---------------------------------------------------------------------------
// Device-wide sense barrier (persistent recurrent kernel, 32 blocks).
// Preceded by a wave-cluster barrier (s_barrier_signal/-wait -3) which is the
// native sync when the dispatch forms a cluster (NOP otherwise).
// ---------------------------------------------------------------------------
__device__ __forceinline__ void grid_barrier(unsigned* cnt, unsigned* phase,
                                             unsigned nblocks, unsigned target) {
  __syncthreads();
  __builtin_amdgcn_s_cluster_barrier();
  if (threadIdx.x == 0) {
    __threadfence();
    unsigned old = __hip_atomic_fetch_add(cnt, 1u, __ATOMIC_ACQ_REL,
                                          __HIP_MEMORY_SCOPE_AGENT);
    if (old == nblocks - 1u) {
      __hip_atomic_store(cnt, 0u, __ATOMIC_RELAXED, __HIP_MEMORY_SCOPE_AGENT);
      __hip_atomic_fetch_add(phase, 1u, __ATOMIC_RELEASE,
                             __HIP_MEMORY_SCOPE_AGENT);
    } else {
      while (__hip_atomic_load(phase, __ATOMIC_ACQUIRE,
                               __HIP_MEMORY_SCOPE_AGENT) < target) {
        __builtin_amdgcn_s_sleep(2);
      }
    }
  }
  __syncthreads();
}

// ---------------------------------------------------------------------------
// Persistent GRU recurrence for one layer.
// Grid = (16 hidden slices of 64 units) x (2 batch chunks of 16). 12 waves.
// Wave w: gate g = w/4, column tile ct = w%4.  Per step:
//   async LDS <- h_prev(bf16) [16 x 1024]; 32x bf16 WMMA over K (grouped in
//   blocks of 4 so loads stay ahead of the matrix pipe); gates -> LDS;
//   fp32 pointwise GRU update; h_new -> global (fp32 + bf16); barrier.
// ---------------------------------------------------------------------------
__global__ __launch_bounds__(384) void gru_recurrent(
    const float* __restrict__ Xg,            // [B*T, 3H] (includes b_ih)
    const unsigned short* __restrict__ Whh,  // [3H, H] bf16
    const float* __restrict__ bhh,           // [3H]
    float* __restrict__ hfp,                 // [2][B][H] fp32 (double buffer)
    unsigned short* __restrict__ hbf,        // [2][B][H] bf16 (double buffer)
    unsigned* bar_cnt, unsigned* bar_phase,
    float* __restrict__ Out, int T)
{
  const int H = 1024, B = 32;
  const int slice = blockIdx.x;              // 0..15 -> 64 hidden units
  const int chunk = blockIdx.y;              // 0..1  -> 16 batches
  const unsigned nblocks = gridDim.x * gridDim.y;

  const int wave  = threadIdx.x >> 5;
  const int lane  = threadIdx.x & 31;
  const int g     = wave >> 2;               // gate 0=r 1=z 2=n
  const int ct    = wave & 3;                // 16-col tile within slice
  const int lhalf = lane >> 4;
  const int ml    = lane & 15;
  const int kb    = lhalf ? 8 : 0;
  const int j0    = slice * 64;

  __shared__ unsigned short Hsh[16 * 1024];  // h_prev tile, bf16, 32 KB
  __shared__ float Gsh[3 * 16 * 64];         // gate pre-activations, 12 KB

  const unsigned hsh_base = (unsigned)(uintptr_t)(&Hsh[0]);
  const int nl = ct * 16 + ml;               // local column 0..63
  const unsigned short* wrow0 = Whh + (size_t)(g * H + j0 + nl) * H;

  for (int t = 0; t < T; ++t) {
    const int rd = t & 1, wr = rd ^ 1;

    // async stage h_prev (bf16) for this batch chunk: 2048 x 16B
    const unsigned short* src = hbf + (size_t)rd * B * H + chunk * 16 * H;
    for (int v = threadIdx.x; v < 2048; v += 384)
      async_copy_b128(hsh_base + (unsigned)v * 16u, src + v * 8);
    wait_async0();
    __syncthreads();

    v8f c = {};
    for (int k0 = 0; k0 < H; k0 += 128) {
      __builtin_prefetch(wrow0 + k0 + 512, 0, 0);
      Frag16 a4[4], b4[4];
#pragma unroll
      for (int u = 0; u < 4; ++u) {
        int kk = k0 + u * 32;
        const uint4* ap = (const uint4*)(Hsh + ml * 1024 + kk + kb);
        a4[u].q[0] = ap[0]; a4[u].q[1] = ap[2];
        const uint4* bp = (const uint4*)(wrow0 + kk + kb);
        b4[u].q[0] = bp[0]; b4[u].q[1] = bp[2];
      }
#pragma unroll
      for (int u = 0; u < 4; ++u)
        c = __builtin_amdgcn_wmma_f32_16x16x32_bf16(
            false, a4[u].v, false, b4[u].v, (short)0, c, false, false);
    }

    // write gate pre-activation tile (with b_hh) to LDS
    {
      float bv = bhh[g * H + j0 + nl];
#pragma unroll
      for (int r = 0; r < 8; ++r) {
        int m = r + lhalf * 8;
        Gsh[g * 1024 + m * 64 + nl] = c[r] + bv;
      }
    }
    __syncthreads();

    // fp32 pointwise GRU update: 16 batches x 64 units
    for (int idx = threadIdx.x; idx < 1024; idx += 384) {
      int m = idx >> 6, j = idx & 63;
      int b = chunk * 16 + m;
      int jg = j0 + j;
      const float* xg = Xg + (size_t)(b * T + t) * 3072;
      float ghr = Gsh[0 * 1024 + m * 64 + j];
      float ghz = Gsh[1 * 1024 + m * 64 + j];
      float ghn = Gsh[2 * 1024 + m * 64 + j];
      float rr = 1.0f / (1.0f + __expf(-(xg[jg]        + ghr)));
      float zz = 1.0f / (1.0f + __expf(-(xg[1024 + jg] + ghz)));
      float nn = tanhf(xg[2048 + jg] + rr * ghn);
      float hp = hfp[(size_t)rd * B * H + (size_t)b * H + jg];
      float hn = (1.0f - zz) * nn + zz * hp;
      hfp[(size_t)wr * B * H + (size_t)b * H + jg] = hn;
      hbf[(size_t)wr * B * H + (size_t)b * H + jg] = f32_to_bf16_rne(hn);
      Out[(size_t)(b * T + t) * H + jg] = hn;
    }

    grid_barrier(bar_cnt, bar_phase, nblocks, (unsigned)(t + 1));
  }
}

// ---------------------------------------------------------------------------
// Host-side orchestration (graph-capture safe: only async ops on `stream`)
// ---------------------------------------------------------------------------
extern "C" void kernel_launch(void* const* d_in, const int* in_sizes, int n_in,
                              void* d_out, int out_size, void* d_ws, size_t ws_size,
                              hipStream_t stream) {
  const int B = 32, T = 512, I = 512, H = 1024, G = 3 * H;  // G = 3072
  const int BT = B * T;                                      // 16384

  const float* x      = (const float*)d_in[0];
  const float* wih_b  = (const float*)d_in[1];
  const float* whh_b  = (const float*)d_in[2];
  const float* bih_b  = (const float*)d_in[3];
  const float* bhh_b  = (const float*)d_in[4];
  const float* wih_r  = (const float*)d_in[5];
  const float* whh_r  = (const float*)d_in[6];
  const float* bih_r  = (const float*)d_in[7];
  const float* bhh_r  = (const float*)d_in[8];
  float* out = (float*)d_out;

  // workspace layout
  char* ws = (char*)d_ws;
  size_t o = 0;
  float* xg = (float*)(ws + o);            o += (size_t)BT * G * sizeof(float);
  unsigned short* inbf  = (unsigned short*)(ws + o); o += (size_t)BT * H * 2;
  unsigned short* wihbf = (unsigned short*)(ws + o); o += (size_t)G * H * 2;
  unsigned short* whhbf = (unsigned short*)(ws + o); o += (size_t)G * H * 2;
  float* hfp = (float*)(ws + o);           o += (size_t)2 * B * H * sizeof(float);
  unsigned short* hbf = (unsigned short*)(ws + o); o += (size_t)2 * B * H * 2;
  unsigned* bar = (unsigned*)(ws + o);     o += 256;
  const size_t hzero_bytes = (size_t)2 * B * H * sizeof(float)
                           + (size_t)2 * B * H * 2 + 256;

  for (int L = 0; L < 3; ++L) {
    const int K = (L == 0) ? I : H;
    const float* src_in = (L == 0) ? x : out + (size_t)(L - 1) * BT * H;
    const float* wih = (L == 0) ? wih_b : wih_r + (size_t)(L - 1) * G * H;
    const float* whh = (L == 0) ? whh_b : whh_r + (size_t)(L - 1) * G * H;
    const float* bih = (L == 0) ? bih_b : bih_r + (size_t)(L - 1) * G;
    const float* bhh = (L == 0) ? bhh_b : bhh_r + (size_t)(L - 1) * G;

    // fp32 -> bf16 conversions for this layer
    cvt_f32_bf16<<<4096, 256, 0, stream>>>(src_in, inbf, (long long)BT * K);
    cvt_f32_bf16<<<4096, 256, 0, stream>>>(wih, wihbf, (long long)G * K);
    cvt_f32_bf16<<<4096, 256, 0, stream>>>(whh, whhbf, (long long)G * H);

    // zero h double-buffers (fp32 + bf16) and barrier state
    hipMemsetAsync(hfp, 0, hzero_bytes, stream);

    // xg = in * w_ih^T + b_ih   (M=16384, N=3072, K=512/1024)
    gemm_xg<<<dim3(G / 64, BT / 128), 256, 0, stream>>>(
        inbf, wihbf, bih, xg, BT, G, K);

    // persistent recurrence over T steps
    gru_recurrent<<<dim3(16, 2), 384, 0, stream>>>(
        xg, whhbf, bhh, hfp, hbf, bar, bar + 1,
        out + (size_t)L * BT * H, T);
  }
}
